// SCAHGTLayer_12403865551349
// MI455X (gfx1250) — compile-verified, matrix operand order checked by reading, not run on hardware
//
#include <hip/hip_runtime.h>
#include <hip/hip_bf16.h>
#include <math.h>

// ---------------- problem constants ----------------
#define N_NODES 50000
#define N_EDGES 5000
#define NNZV    250000
#define IN_DIMC 256
#define OCC     512
#define FFC     2048
#define NPAD    50048   // 391 * 128
#define EPAD    5120    // 40  * 128
#define EPSBN   1e-5f
#define LDAK    40      // LDS k-stride (pad 32 -> 40 halfwords = 80B; 16B aligned, conflict-free)

typedef unsigned short u16;
typedef unsigned int   u32;
typedef __attribute__((ext_vector_type(16))) __bf16 v16bf;
typedef __attribute__((ext_vector_type(8)))  float  v8f;

union BF8 { uint4 u; __bf16 h[8]; u16 s[8]; };

__device__ __forceinline__ u16 f2bf(float f) {
  u32 u = __float_as_uint(f);
  u += 0x7fffu + ((u >> 16) & 1u);           // round to nearest even
  return (u16)(u >> 16);
}
__device__ __forceinline__ u32 f2ord(float f) {  // monotonic float->uint map
  u32 u = __float_as_uint(f);
  return (u & 0x80000000u) ? ~u : (u | 0x80000000u);
}
__device__ __forceinline__ float ord2f(u32 o) {
  u32 u = (o & 0x80000000u) ? (o & 0x7fffffffu) : ~o;
  return __uint_as_float(u);
}

// gfx1250 async global->LDS copy (ASYNCcnt-tracked), per-lane addresses
__device__ __forceinline__ void async_ld_b128(unsigned ldsoff, const void* gptr) {
  asm volatile("global_load_async_to_lds_b128 %0, %1, off"
               :: "v"(ldsoff), "v"((unsigned long long)(size_t)gptr)
               : "memory");
}
__device__ __forceinline__ void wait_async0() {
  asm volatile("s_wait_asynccnt 0x0" ::: "memory");
}

// ---------------- zero fill ----------------
__global__ void zero_u32_kernel(u32* __restrict__ p, long long n) {
  long long i = (long long)blockIdx.x * 256 + threadIdx.x;
  if (i < n) p[i] = 0u;
}

// ---------------- f32 -> bf16 with zero padding ----------------
__global__ void cvt_bf16_kernel(const float* __restrict__ src, u16* __restrict__ dst,
                                long long realElems, long long totalElems) {
  long long i = (long long)blockIdx.x * 256 + threadIdx.x;
  if (i >= totalElems) return;
  dst[i] = (i < realElems) ? f2bf(src[i]) : (u16)0;
}

// ---------------- f32 [K,N] -> bf16 transposed [N,K] (weights) ----------------
__global__ void cvt_bf16_t_kernel(const float* __restrict__ src, u16* __restrict__ dst,
                                  int K, int N) {
  long long i = (long long)blockIdx.x * 256 + threadIdx.x;
  if (i >= (long long)K * N) return;
  int n = (int)(i / K), kk = (int)(i % K);
  dst[i] = f2bf(src[(long long)kk * N + n]);
}

// ---------------- bf16 WMMA GEMM: C[M,N] = A[M,K] @ B[K,N] (+bias)(+add)(gelu) --------
// A row-major [M,K]; BT = B transposed, row-major [N,K]. M % 128 == 0 (zero padded),
// N % 128 == 0, K % 32 == 0. 256 threads = 8 waves; 128x128 C tile; each wave owns
// 32(M) x 64(N): 2 A-frags x 4 B-frags -> 8 v_wmma_f32_16x16x32_bf16 per K-step.
// Global->LDS via global_load_async_to_lds_b128 with double-buffered LDS so the next
// tile's copies overlap the current tile's WMMAs.
template<bool BIAS, bool ADD, bool GELU_ACT, bool STF32, bool STBF>
__global__ __launch_bounds__(256) void gemm_bf16_kernel(
    const u16* __restrict__ A, const u16* __restrict__ BT,
    const float* __restrict__ bias, const float* __restrict__ addsrc,
    float* __restrict__ C, u16* __restrict__ Cb, int N, int K)
{
  __shared__ u16 As[2][128][LDAK];
  __shared__ u16 Bs[2][128][LDAK];

  const int tid  = threadIdx.x;
  const int lane = tid & 31;
  const int wid  = tid >> 5;
  const int wm   = wid >> 1;       // 0..3 : 32-row group
  const int wn   = wid & 1;        // 0..1 : 64-col group
  const int bm   = blockIdx.y << 7;
  const int bn   = blockIdx.x << 7;

  // copy mapping: each thread owns 16 bf16 (2 x b128) of one row in each tile
  const int lr = tid >> 1;             // 0..127
  const int lc = (tid & 1) << 4;       // 0 or 16

  const u16* Arow = A  + (long long)(bm + lr) * K + lc;
  const u16* Brow = BT + (long long)(bn + lr) * K + lc;
  const unsigned ldsA = (unsigned)(size_t)&As[0][lr][lc];
  const unsigned ldsB = (unsigned)(size_t)&Bs[0][lr][lc];
  const unsigned bufStride = (unsigned)(128 * LDAK * 2);

  // fragment lane mapping (ISA 7.12.2, 16-bit 16x32):
  // lanes 0-15: elems 0..7 = K kb..kb+7, elems 8..15 = K kb+16..kb+23, kb=0; lanes 16-31: kb=8
  const int r  = lane & 15;
  const int kb = (lane >> 4) << 3;

  v8f acc[2][4] = {};

  const int nT = K >> 5;
  // prologue: tile 0
  {
    async_ld_b128(ldsA,      Arow);
    async_ld_b128(ldsA + 16, Arow + 8);
    async_ld_b128(ldsB,      Brow);
    async_ld_b128(ldsB + 16, Brow + 8);
  }

  for (int t = 0; t < nT; ++t) {
    const int cur = t & 1;
    wait_async0();          // my async writes into LDS done
    __syncthreads();        // everyone's writes done; everyone done reading other buf
    if (t + 1 < nT) {
      const unsigned boff = (cur ^ 1) ? bufStride : 0u;
      const int kt = (t + 1) << 5;
      async_ld_b128(ldsA + boff,      Arow + kt);
      async_ld_b128(ldsA + boff + 16, Arow + kt + 8);
      async_ld_b128(ldsB + boff,      Brow + kt);
      async_ld_b128(ldsB + boff + 16, Brow + kt + 8);
    }

    BF8 lo, hi;
    v16bf af[2], bfr[4];
    #pragma unroll
    for (int sub = 0; sub < 2; ++sub) {
      lo.u = *(const uint4*)(&As[cur][wm * 32 + sub * 16 + r][kb]);
      hi.u = *(const uint4*)(&As[cur][wm * 32 + sub * 16 + r][kb + 16]);
      #pragma unroll
      for (int j = 0; j < 8; ++j) { af[sub][j] = lo.h[j]; af[sub][j + 8] = hi.h[j]; }
    }
    #pragma unroll
    for (int f = 0; f < 4; ++f) {
      lo.u = *(const uint4*)(&Bs[cur][wn * 64 + f * 16 + r][kb]);
      hi.u = *(const uint4*)(&Bs[cur][wn * 64 + f * 16 + r][kb + 16]);
      #pragma unroll
      for (int j = 0; j < 8; ++j) { bfr[f][j] = lo.h[j]; bfr[f][j + 8] = hi.h[j]; }
    }
    #pragma unroll
    for (int sub = 0; sub < 2; ++sub)
      #pragma unroll
      for (int f = 0; f < 4; ++f)
        acc[sub][f] = __builtin_amdgcn_wmma_f32_16x16x32_bf16(
            false, af[sub], false, bfr[f], (short)0, acc[sub][f], false, false);
  }

  // C layout (ISA 7.12.2): VGPR i -> M=i (lanes 0-15) / M=i+8 (lanes 16-31), N=lane&15
  const int row0 = bm + wm * 32 + ((lane >> 4) << 3);
  const int col0 = bn + wn * 64 + (lane & 15);
  #pragma unroll
  for (int sub = 0; sub < 2; ++sub) {
    #pragma unroll
    for (int f = 0; f < 4; ++f) {
      const int col = col0 + f * 16;
      #pragma unroll
      for (int i = 0; i < 8; ++i) {
        long long row = row0 + sub * 16 + i;
        float v = acc[sub][f][i];
        if (BIAS) v += bias[col];
        if (ADD)  v += addsrc[row * N + col];
        if (GELU_ACT) v = 0.5f * v * (1.0f + erff(v * 0.70710678118f));
        if (STF32) C[row * N + col] = v;
        if (STBF)  Cb[row * N + col] = f2bf(v);
      }
    }
  }
}

// ---------------- attention: per-(nnz,head) dot + segment max ----------------
__global__ void attn_score_kernel(const float* __restrict__ Q, const float* __restrict__ Km,
                                  const int* __restrict__ qid, const int* __restrict__ kid,
                                  const int* __restrict__ seg, float* __restrict__ s,
                                  u32* __restrict__ segmax, int nnz) {
  long long g = (long long)blockIdx.x * 256 + threadIdx.x;
  if (g >= (long long)nnz * 8) return;
  int i = (int)(g >> 3), h = (int)(g & 7);
  const float4* qp = (const float4*)(Q + (long long)qid[i] * OCC + h * 64);
  const float4* kp = (const float4*)(Km + (long long)kid[i] * OCC + h * 64);
  float acc = 0.f;
  #pragma unroll
  for (int j = 0; j < 16; ++j) {
    float4 a = qp[j], b = kp[j];
    acc += a.x * b.x + a.y * b.y + a.z * b.z + a.w * b.w;
  }
  acc *= 0.125f;  // 1/sqrt(64)
  s[g] = acc;
  atomicMax(&segmax[(long long)seg[i] * 8 + h], f2ord(acc));
}

// ---------------- attention: w = exp(s - max), segment sum ----------------
__global__ void attn_expsum_kernel(float* __restrict__ s, const u32* __restrict__ segmax,
                                   float* __restrict__ segsum, const int* __restrict__ seg,
                                   int nnz) {
  long long g = (long long)blockIdx.x * 256 + threadIdx.x;
  if (g >= (long long)nnz * 8) return;
  int i = (int)(g >> 3), h = (int)(g & 7);
  float m = ord2f(segmax[(long long)seg[i] * 8 + h]);
  float w = expf(s[g] - m);
  s[g] = w;
  atomicAdd(&segsum[(long long)seg[i] * 8 + h], w);
}

// ---------------- attention: out[seg] += V[vid] * (w / segsum[seg]) ----------------
__global__ void attn_accum_kernel(const float* __restrict__ s, const float* __restrict__ segsum,
                                  const float* __restrict__ V, const int* __restrict__ vid,
                                  const int* __restrict__ seg, float* __restrict__ out, int nnz) {
  long long g = (long long)blockIdx.x * 256 + threadIdx.x;
  if (g >= (long long)nnz * 8) return;
  int i = (int)(g >> 3), h = (int)(g & 7);
  int sg = seg[i];
  float coef = s[g] / segsum[(long long)sg * 8 + h];
  const float4* vp = (const float4*)(V + (long long)vid[i] * OCC + h * 64);
  float* op = out + (long long)sg * OCC + h * 64;
  #pragma unroll
  for (int j = 0; j < 16; ++j) {
    float4 v = vp[j];
    atomicAdd(&op[4 * j + 0], v.x * coef);
    atomicAdd(&op[4 * j + 1], v.y * coef);
    atomicAdd(&op[4 * j + 2], v.z * coef);
    atomicAdd(&op[4 * j + 3], v.w * coef);
  }
}

// ---------------- batchnorm column stats (sum, sumsq) ----------------
__global__ void bn_stats_kernel(const float* __restrict__ x, float* __restrict__ st,
                                int rows, int C) {
  int c  = blockIdx.x * 256 + threadIdx.x;
  int r0 = blockIdx.y * 128;
  int re = min(r0 + 128, rows);
  float s = 0.f, q = 0.f;
  for (int rr = r0; rr < re; ++rr) {
    float v = x[(long long)rr * C + c];
    s += v; q += v * v;
  }
  atomicAdd(&st[c], s);
  atomicAdd(&st[C + c], q);
}

// normalize in place + bf16 copy (padding rows -> bf16 zero)
__global__ void bn_apply_mid_kernel(float* __restrict__ x, const float* __restrict__ st,
                                    const float* __restrict__ g, const float* __restrict__ b,
                                    u16* __restrict__ xbf, int rows, int rowsPad, int C,
                                    float invn) {
  long long idx = (long long)blockIdx.x * 256 + threadIdx.x;
  if (idx >= (long long)rowsPad * C) return;
  int c = (int)(idx % C);
  long long r = idx / C;
  if (r < rows) {
    float mu  = st[c] * invn;
    float var = st[C + c] * invn - mu * mu;
    float v   = (x[idx] - mu) * rsqrtf(var + EPSBN) * g[c] + b[c];
    x[idx] = v;
    xbf[idx] = f2bf(v);
  } else {
    xbf[idx] = (u16)0;
  }
}

// normalize -> final fp32 output (unpadded)
__global__ void bn_apply_out_kernel(const float* __restrict__ y, const float* __restrict__ st,
                                    const float* __restrict__ g, const float* __restrict__ b,
                                    float* __restrict__ out, int rows, int C, float invn) {
  long long idx = (long long)blockIdx.x * 256 + threadIdx.x;
  if (idx >= (long long)rows * C) return;
  int c = (int)(idx % C);
  float mu  = st[c] * invn;
  float var = st[C + c] * invn - mu * mu;
  out[idx] = (y[idx] - mu) * rsqrtf(var + EPSBN) * g[c] + b[c];
}

// ---------------- launcher ----------------
static inline int blks(long long n) { return (int)((n + 255) / 256); }

extern "C" void kernel_launch(void* const* d_in, const int* in_sizes, int n_in,
                              void* d_out, int out_size, void* d_ws, size_t ws_size,
                              hipStream_t stream) {
  (void)in_sizes; (void)n_in; (void)out_size; (void)ws_size;

  const float* q   = (const float*)d_in[0];
  const float* k   = (const float*)d_in[1];
  const float* ef  = (const float*)d_in[3];
  const float* Wq1 = (const float*)d_in[4];   // Wq_n2h 256x512
  const float* Wk1 = (const float*)d_in[5];   // Wk_n2h 256x512
  const float* Wv1 = (const float*)d_in[6];   // Wv_n2h 256x512
  const float* Wo1 = (const float*)d_in[7];   // Wo_n2h 512x512
  const float* We  = (const float*)d_in[8];   // 256x256
  const float* Wq2 = (const float*)d_in[9];   // Wq_h2n 256x512
  const float* Wk2 = (const float*)d_in[10];  // 512x512
  const float* Wv2 = (const float*)d_in[11];  // 512x512
  const float* Wo2 = (const float*)d_in[12];  // 512x512
  const float* W1  = (const float*)d_in[13];  // 512x2048
  const float* b1  = (const float*)d_in[14];
  const float* W2  = (const float*)d_in[15];  // 2048x512
  const float* b2  = (const float*)d_in[16];
  const float* Wr  = (const float*)d_in[17];  // Wres 256x512
  const float* g1  = (const float*)d_in[18];
  const float* be1 = (const float*)d_in[19];
  const float* g2  = (const float*)d_in[20];
  const float* be2 = (const float*)d_in[21];
  const int* nid   = (const int*)d_in[22];
  const int* eid   = (const int*)d_in[23];
  float* out       = (float*)d_out;

  // workspace bump allocator (256B aligned)
  size_t off = 0;
  auto alloc = [&](size_t bytes) -> void* {
    off = (off + 255) & ~(size_t)255;
    void* p = (char*)d_ws + off;
    off += bytes;
    return p;
  };

  u16* qbf   = (u16*)alloc((size_t)NPAD * IN_DIMC * 2);
  u16* kbf   = (u16*)alloc((size_t)NPAD * IN_DIMC * 2);
  u16* efbf  = (u16*)alloc((size_t)EPAD * IN_DIMC * 2);
  u16* einbf = (u16*)alloc((size_t)EPAD * IN_DIMC * 2);
  // weights stored transposed: [N,K] bf16
  u16* wWq1  = (u16*)alloc((size_t)IN_DIMC * OCC * 2);
  u16* wWk1  = (u16*)alloc((size_t)IN_DIMC * OCC * 2);
  u16* wWv1  = (u16*)alloc((size_t)IN_DIMC * OCC * 2);
  u16* wWo1  = (u16*)alloc((size_t)OCC * OCC * 2);
  u16* wWe   = (u16*)alloc((size_t)IN_DIMC * IN_DIMC * 2);
  u16* wWq2  = (u16*)alloc((size_t)IN_DIMC * OCC * 2);
  u16* wWk2  = (u16*)alloc((size_t)OCC * OCC * 2);
  u16* wWv2  = (u16*)alloc((size_t)OCC * OCC * 2);
  u16* wWo2  = (u16*)alloc((size_t)OCC * OCC * 2);
  u16* wW1   = (u16*)alloc((size_t)OCC * FFC * 2);
  u16* wW2   = (u16*)alloc((size_t)FFC * OCC * 2);
  u16* wWr   = (u16*)alloc((size_t)IN_DIMC * OCC * 2);
  float* bufA = (float*)alloc((size_t)NPAD * OCC * 4);   // Q1 -> Q2 -> y
  float* bufB = (float*)alloc((size_t)NPAD * OCC * 4);   // V1 -> nu
  float* bufC = (float*)alloc((size_t)NPAD * OCC * 4);   // nuo -> x
  float* K1f  = (float*)alloc((size_t)EPAD * OCC * 4);
  float* hef  = (float*)alloc((size_t)EPAD * OCC * 4);
  u16*  hebf  = (u16*)alloc((size_t)EPAD * OCC * 2);
  u16*  heobf = (u16*)alloc((size_t)EPAD * OCC * 2);
  float* K2f  = (float*)alloc((size_t)EPAD * OCC * 4);
  float* V2f  = (float*)alloc((size_t)EPAD * OCC * 4);
  float* s1   = (float*)alloc((size_t)NNZV * 8 * 4);
  float* s2   = (float*)alloc((size_t)NNZV * 8 * 4);
  u32*  emax  = (u32*)alloc((size_t)EPAD * 8 * 4);
  float* esum = (float*)alloc((size_t)EPAD * 8 * 4);
  u32*  nmax  = (u32*)alloc((size_t)NPAD * 8 * 4);
  float* nsum = (float*)alloc((size_t)NPAD * 8 * 4);
  u16*  nubf  = (u16*)alloc((size_t)NPAD * OCC * 2);
  u16*  xbf   = (u16*)alloc((size_t)NPAD * OCC * 2);
  u16*  hbf   = (u16*)alloc((size_t)NPAD * FFC * 2);
  float* bn1  = (float*)alloc((size_t)2 * OCC * 4);
  float* bn2  = (float*)alloc((size_t)2 * OCC * 4);

  auto zero = [&](void* p, size_t bytes) {
    long long n = (long long)(bytes / 4);
    zero_u32_kernel<<<blks(n), 256, 0, stream>>>((u32*)p, n);
  };
  auto cvt = [&](const float* src, u16* dst, long long real, long long total) {
    cvt_bf16_kernel<<<blks(total), 256, 0, stream>>>(src, dst, real, total);
  };
  auto cvtT = [&](const float* src, u16* dst, int K, int N) {  // [K,N] -> [N,K]
    cvt_bf16_t_kernel<<<blks((long long)K * N), 256, 0, stream>>>(src, dst, K, N);
  };

  // ---- convert activations (zero-padded) and weights (transposed) to bf16 ----
  cvt(q,  qbf,  (long long)N_NODES * IN_DIMC, (long long)NPAD * IN_DIMC);
  cvt(k,  kbf,  (long long)N_NODES * IN_DIMC, (long long)NPAD * IN_DIMC);
  cvt(ef, efbf, (long long)N_EDGES * IN_DIMC, (long long)EPAD * IN_DIMC);
  cvtT(Wq1, wWq1, IN_DIMC, OCC);
  cvtT(Wk1, wWk1, IN_DIMC, OCC);
  cvtT(Wv1, wWv1, IN_DIMC, OCC);
  cvtT(Wo1, wWo1, OCC, OCC);
  cvtT(We,  wWe,  IN_DIMC, IN_DIMC);
  cvtT(Wq2, wWq2, IN_DIMC, OCC);
  cvtT(Wk2, wWk2, OCC, OCC);
  cvtT(Wv2, wWv2, OCC, OCC);
  cvtT(Wo2, wWo2, OCC, OCC);
  cvtT(W1,  wW1,  OCC, FFC);
  cvtT(W2,  wW2,  FFC, OCC);
  cvtT(Wr,  wWr,  IN_DIMC, OCC);

  dim3 blk(256);
  // ---- phase A: node -> hyperedge attention ----
  gemm_bf16_kernel<false,false,false,false,true>
      <<<dim3(IN_DIMC/128, EPAD/128), blk, 0, stream>>>(efbf, wWe, nullptr, nullptr,
                                                        nullptr, einbf, IN_DIMC, IN_DIMC);
  gemm_bf16_kernel<false,false,false,true,false>
      <<<dim3(OCC/128, NPAD/128), blk, 0, stream>>>(kbf, wWq1, nullptr, nullptr,
                                                    bufA, nullptr, OCC, IN_DIMC);
  gemm_bf16_kernel<false,false,false,true,false>
      <<<dim3(OCC/128, NPAD/128), blk, 0, stream>>>(kbf, wWv1, nullptr, nullptr,
                                                    bufB, nullptr, OCC, IN_DIMC);
  gemm_bf16_kernel<false,false,false,true,false>
      <<<dim3(OCC/128, EPAD/128), blk, 0, stream>>>(einbf, wWk1, nullptr, nullptr,
                                                    K1f, nullptr, OCC, IN_DIMC);

  zero(emax, (size_t)EPAD * 8 * 4);
  zero(esum, (size_t)EPAD * 8 * 4);
  attn_score_kernel<<<blks((long long)NNZV * 8), blk, 0, stream>>>(
      bufA, K1f, nid, eid, eid, s1, emax, NNZV);
  attn_expsum_kernel<<<blks((long long)NNZV * 8), blk, 0, stream>>>(
      s1, emax, esum, eid, NNZV);
  zero(hef, (size_t)EPAD * OCC * 4);
  attn_accum_kernel<<<blks((long long)NNZV * 8), blk, 0, stream>>>(
      s1, esum, bufB, nid, eid, hef, NNZV);

  cvt(hef, hebf, (long long)EPAD * OCC, (long long)EPAD * OCC);
  gemm_bf16_kernel<false,false,false,false,true>
      <<<dim3(OCC/128, EPAD/128), blk, 0, stream>>>(hebf, wWo1, nullptr, nullptr,
                                                    nullptr, heobf, OCC, OCC);

  // ---- phase B: hyperedge -> node attention ----
  gemm_bf16_kernel<false,false,false,true,false>
      <<<dim3(OCC/128, NPAD/128), blk, 0, stream>>>(qbf, wWq2, nullptr, nullptr,
                                                    bufA, nullptr, OCC, IN_DIMC);
  gemm_bf16_kernel<false,false,false,true,false>
      <<<dim3(OCC/128, EPAD/128), blk, 0, stream>>>(heobf, wWk2, nullptr, nullptr,
                                                    K2f, nullptr, OCC, OCC);
  gemm_bf16_kernel<false,false,false,true,false>
      <<<dim3(OCC/128, EPAD/128), blk, 0, stream>>>(heobf, wWv2, nullptr, nullptr,
                                                    V2f, nullptr, OCC, OCC);

  zero(nmax, (size_t)NPAD * 8 * 4);
  zero(nsum, (size_t)NPAD * 8 * 4);
  attn_score_kernel<<<blks((long long)NNZV * 8), blk, 0, stream>>>(
      bufA, K2f, nid, eid, nid, s2, nmax, NNZV);
  attn_expsum_kernel<<<blks((long long)NNZV * 8), blk, 0, stream>>>(
      s2, nmax, nsum, nid, NNZV);
  zero(bufB, (size_t)NPAD * OCC * 4);     // nu accumulator
  attn_accum_kernel<<<blks((long long)NNZV * 8), blk, 0, stream>>>(
      s2, nsum, V2f, eid, nid, bufB, NNZV);

  // nuo = nu @ Wo_h2n ; x = nuo + q @ Wres
  cvt(bufB, nubf, (long long)NPAD * OCC, (long long)NPAD * OCC);
  gemm_bf16_kernel<false,false,false,true,false>
      <<<dim3(OCC/128, NPAD/128), blk, 0, stream>>>(nubf, wWo2, nullptr, nullptr,
                                                    bufC, nullptr, OCC, OCC);
  gemm_bf16_kernel<false,true,false,true,false>
      <<<dim3(OCC/128, NPAD/128), blk, 0, stream>>>(qbf, wWr, nullptr, bufC,
                                                    bufC, nullptr, OCC, IN_DIMC);

  // ---- BN1 -> FFN -> BN2 ----
  zero(bn1, (size_t)2 * OCC * 4);
  bn_stats_kernel<<<dim3(OCC/256, (N_NODES + 127) / 128), blk, 0, stream>>>(
      bufC, bn1, N_NODES, OCC);
  bn_apply_mid_kernel<<<blks((long long)NPAD * OCC), blk, 0, stream>>>(
      bufC, bn1, g1, be1, xbf, N_NODES, NPAD, OCC, 1.0f / N_NODES);

  // h = gelu(x @ W1 + b1)   (bf16 out)
  gemm_bf16_kernel<true,false,true,false,true>
      <<<dim3(FFC/128, NPAD/128), blk, 0, stream>>>(xbf, wW1, b1, nullptr,
                                                    nullptr, hbf, FFC, OCC);
  // y = h @ W2 + b2 + x
  gemm_bf16_kernel<true,true,false,true,false>
      <<<dim3(OCC/128, NPAD/128), blk, 0, stream>>>(hbf, wW2, b2, bufC,
                                                    bufA, nullptr, OCC, FFC);

  zero(bn2, (size_t)2 * OCC * 4);
  bn_stats_kernel<<<dim3(OCC/256, (N_NODES + 127) / 128), blk, 0, stream>>>(
      bufA, bn2, N_NODES, OCC);
  bn_apply_out_kernel<<<blks((long long)N_NODES * OCC), blk, 0, stream>>>(
      bufA, bn2, g2, be2, out, N_NODES, OCC, 1.0f / N_NODES);
}
